// LSTM_Head_90194313216269
// MI455X (gfx1250) — matrix operand order repaired and checked by reading, not compiled, not run on hardware
//
#include <hip/hip_runtime.h>
#include <hip/hip_bf16.h>

#define B_SZ 512
#define T_SZ 1024
#define F_SZ 128
#define H_SZ 64
#define G_SZ 256   // 4*H

typedef __attribute__((ext_vector_type(16))) __bf16 v16bf;
typedef __attribute__((ext_vector_type(8)))  __bf16 v8bf;
typedef __attribute__((ext_vector_type(8)))  float  v8f;

__device__ __forceinline__ float sigm(float x) { return 1.0f / (1.0f + __expf(-x)); }
__device__ __forceinline__ float tanh_fast(float x) { return 1.0f - 2.0f / (1.0f + __expf(2.0f * x)); }

__device__ __forceinline__ v8f wmma_bf16(v16bf a, v16bf b, v8f c) {
  // (neg_a, A, neg_b, B, c_mod, C, reuse_a, reuse_b)
  return __builtin_amdgcn_wmma_f32_16x16x32_bf16(false, a, false, b, (short)0, c, false, false);
}

// A tile: 16 (M) x 32 (K) bf16 from row-major [16][ld] LDS buffer, K offset kb.
// Per ISA 7.12.2: lane holds row m=lane&15; elements 0..7 -> K = half*8 + 0..7,
// elements 8..15 -> K = 16 + half*8 + 0..7 (half = lane>>4).
__device__ __forceinline__ v16bf load_a_tile(const __bf16* base, int ld, int kb, int lane) {
  const int m = lane & 15;
  const int half = lane >> 4;
  const __bf16* row = base + m * ld + kb + half * 8;
  v8bf lo = *(const v8bf*)(row);        // K = kb + half*8 .. +7   (16B aligned)
  v8bf hi = *(const v8bf*)(row + 16);   // K = kb + 16 + half*8 .. +7
  v16bf a;
#pragma unroll
  for (int i = 0; i < 8; ++i) { a[i] = lo[i]; a[i + 8] = hi[i]; }
  return a;
}

// B tile: 32 (K) x 16 (N) bf16 from row-major [K][ldn] LDS buffer.
// Lane holds column n=lane&15; elements 0..7 -> K = half*8 + 0..7, elements 8..15 -> K = 16 + half*8 + 0..7.
__device__ __forceinline__ v16bf load_b_tile(const __bf16* base, int ldn, int kb, int nb, int lane) {
  const int n = nb * 16 + (lane & 15);
  const int half = lane >> 4;
  v16bf b;
#pragma unroll
  for (int i = 0; i < 8; ++i) {
    b[i] = base[(size_t)(kb + half * 8 + i) * ldn + n];
    b[i + 8] = base[(size_t)(kb + 16 + half * 8 + i) * ldn + n];
  }
  return b;
}

// ---------------- Encoder: fused x@W projection + LSTM scan ----------------
__global__ __launch_bounds__(256)
void lstm_encoder_kernel(const float* __restrict__ x,
                         const float* __restrict__ enc_W,
                         const float* __restrict__ enc_U,
                         const float* __restrict__ enc_b,
                         float* __restrict__ hT_out) {
  __shared__ __bf16 Wl[F_SZ * G_SZ];   // 64 KB
  __shared__ __bf16 Ul[H_SZ * G_SZ];   // 32 KB
  __shared__ float  bl[G_SZ];
  __shared__ __bf16 Xl[16 * F_SZ];     // x tile (bf16)
  __shared__ __bf16 Hl[16 * H_SZ];     // h carry (bf16, WMMA A source)
  __shared__ float  Cl[16 * H_SZ];     // c carry (f32)
  __shared__ float  Zl[16 * G_SZ];     // gate pre-activations (f32)

  const int tid  = threadIdx.x;
  const int wave = tid >> 5;
  const int lane = tid & 31;
  const int b0   = blockIdx.x * 16;

  for (int i = tid; i < F_SZ * G_SZ; i += 256) Wl[i] = (__bf16)enc_W[i];
  for (int i = tid; i < H_SZ * G_SZ; i += 256) Ul[i] = (__bf16)enc_U[i];
  if (tid < G_SZ) bl[tid] = enc_b[tid];
  for (int i = tid; i < 16 * H_SZ; i += 256) { Hl[i] = (__bf16)0.0f; Cl[i] = 0.0f; }
  __syncthreads();

  for (int t = 0; t < T_SZ; ++t) {
    // stage x[b0..b0+15, t, :] -> bf16 LDS (coalesced 128-wide rows)
    for (int i = tid; i < 16 * F_SZ; i += 256) {
      int m = i >> 7, f = i & (F_SZ - 1);
      Xl[i] = (__bf16)x[((size_t)(b0 + m) * T_SZ + t) * F_SZ + f];
    }
    __syncthreads();

    // z = X@W + H@U   (16 x 256, 16 N-tiles; 2 per wave)
#pragma unroll
    for (int j = 0; j < 2; ++j) {
      const int nb = 2 * wave + j;
      v8f acc = {};
#pragma unroll
      for (int kb = 0; kb < F_SZ; kb += 32)
        acc = wmma_bf16(load_a_tile(Xl, F_SZ, kb, lane), load_b_tile(Wl, G_SZ, kb, nb, lane), acc);
#pragma unroll
      for (int kb = 0; kb < H_SZ; kb += 32)
        acc = wmma_bf16(load_a_tile(Hl, H_SZ, kb, lane), load_b_tile(Ul, G_SZ, kb, nb, lane), acc);
      const int n = nb * 16 + (lane & 15);
      const int mhalf = (lane >> 4) * 8;
#pragma unroll
      for (int r = 0; r < 8; ++r) Zl[(r + mhalf) * G_SZ + n] = acc[r];
    }
    __syncthreads();

    // gates (Keras order i,f,g,o), f32 state update
    for (int i = tid; i < 16 * H_SZ; i += 256) {
      int m = i >> 6, j2 = i & (H_SZ - 1);
      float zi = Zl[m * G_SZ + j2]              + bl[j2];
      float zf = Zl[m * G_SZ + H_SZ + j2]       + bl[H_SZ + j2];
      float zg = Zl[m * G_SZ + 2 * H_SZ + j2]   + bl[2 * H_SZ + j2];
      float zo = Zl[m * G_SZ + 3 * H_SZ + j2]   + bl[3 * H_SZ + j2];
      float c  = sigm(zf) * Cl[i] + sigm(zi) * tanh_fast(zg);
      float h  = sigm(zo) * tanh_fast(c);
      Cl[i] = c;
      Hl[i] = (__bf16)h;
      if (t == T_SZ - 1) hT_out[(size_t)(b0 + m) * H_SZ + j2] = h;
    }
    __syncthreads();
  }
}

// -------- Decoder: constant input projection + scan + fused dense head --------
__global__ __launch_bounds__(256)
void lstm_decoder_kernel(const float* __restrict__ hT,
                         const float* __restrict__ dec_W,
                         const float* __restrict__ dec_U,
                         const float* __restrict__ dec_b,
                         const float* __restrict__ dense_W,
                         const float* __restrict__ dense_b,
                         float* __restrict__ out) {
  __shared__ __bf16 Ul[H_SZ * G_SZ];     // 32 KB
  __shared__ __bf16 Wdl[H_SZ * G_SZ];    // 32 KB
  __shared__ __bf16 Dl[H_SZ * F_SZ];     // 16 KB dense_W
  __shared__ float  dpl[16 * G_SZ];      // constant per-step input projection
  __shared__ float  Zl[16 * G_SZ];
  __shared__ __bf16 Hl[16 * H_SZ];
  __shared__ __bf16 He[16 * H_SZ];       // encoded (bf16 A tile)
  __shared__ float  Cl[16 * H_SZ];
  __shared__ float  dbl[F_SZ];

  const int tid  = threadIdx.x;
  const int wave = tid >> 5;
  const int lane = tid & 31;
  const int b0   = blockIdx.x * 16;

  for (int i = tid; i < H_SZ * G_SZ; i += 256) { Ul[i] = (__bf16)dec_U[i]; Wdl[i] = (__bf16)dec_W[i]; }
  for (int i = tid; i < H_SZ * F_SZ; i += 256) Dl[i] = (__bf16)dense_W[i];
  if (tid < F_SZ) dbl[tid] = dense_b[tid];
  for (int i = tid; i < 16 * H_SZ; i += 256) {
    He[i] = (__bf16)hT[(size_t)(b0 + (i >> 6)) * H_SZ + (i & (H_SZ - 1))];
    Hl[i] = (__bf16)0.0f;
    Cl[i] = 0.0f;
  }
  __syncthreads();

  // dp = encoded @ dec_W + dec_b  (constant over all timesteps)
#pragma unroll
  for (int j = 0; j < 2; ++j) {
    const int nb = 2 * wave + j;
    const int n = nb * 16 + (lane & 15);
    const float bn = dec_b[n];
    v8f acc;
#pragma unroll
    for (int r = 0; r < 8; ++r) acc[r] = bn;
#pragma unroll
    for (int kb = 0; kb < H_SZ; kb += 32)
      acc = wmma_bf16(load_a_tile(He, H_SZ, kb, lane), load_b_tile(Wdl, G_SZ, kb, nb, lane), acc);
    const int mhalf = (lane >> 4) * 8;
#pragma unroll
    for (int r = 0; r < 8; ++r) dpl[(r + mhalf) * G_SZ + n] = acc[r];
  }
  __syncthreads();

  for (int t = 0; t < T_SZ; ++t) {
    // z = dp + H@U
#pragma unroll
    for (int j = 0; j < 2; ++j) {
      const int nb = 2 * wave + j;
      const int n = nb * 16 + (lane & 15);
      const int mhalf = (lane >> 4) * 8;
      v8f acc;
#pragma unroll
      for (int r = 0; r < 8; ++r) acc[r] = dpl[(r + mhalf) * G_SZ + n];
#pragma unroll
      for (int kb = 0; kb < H_SZ; kb += 32)
        acc = wmma_bf16(load_a_tile(Hl, H_SZ, kb, lane), load_b_tile(Ul, G_SZ, kb, nb, lane), acc);
#pragma unroll
      for (int r = 0; r < 8; ++r) Zl[(r + mhalf) * G_SZ + n] = acc[r];
    }
    __syncthreads();

    // gates (dec_b already folded into dp)
    for (int i = tid; i < 16 * H_SZ; i += 256) {
      int m = i >> 6, j2 = i & (H_SZ - 1);
      float zi = Zl[m * G_SZ + j2];
      float zf = Zl[m * G_SZ + H_SZ + j2];
      float zg = Zl[m * G_SZ + 2 * H_SZ + j2];
      float zo = Zl[m * G_SZ + 3 * H_SZ + j2];
      float c  = sigm(zf) * Cl[i] + sigm(zi) * tanh_fast(zg);
      float h  = sigm(zo) * tanh_fast(c);
      Cl[i] = c;
      Hl[i] = (__bf16)h;
    }
    __syncthreads();

    // fused dense head: out[b, t, :] = sigmoid(h @ dense_W + dense_b)
    {
      const int nb = wave;                  // 8 waves x 16 = F = 128
      const int n = nb * 16 + (lane & 15);
      const float bn = dbl[n];
      v8f acc;
#pragma unroll
      for (int r = 0; r < 8; ++r) acc[r] = bn;
#pragma unroll
      for (int kb = 0; kb < H_SZ; kb += 32)
        acc = wmma_bf16(load_a_tile(Hl, H_SZ, kb, lane), load_b_tile(Dl, F_SZ, kb, nb, lane), acc);
      const int mhalf = (lane >> 4) * 8;
#pragma unroll
      for (int r = 0; r < 8; ++r) {
        const int b = b0 + r + mhalf;
        out[((size_t)b * T_SZ + t) * F_SZ + n] = sigm(acc[r]);
      }
    }
    // next write to Hl/Zl is beyond the next __syncthreads in the following iter
  }
}

extern "C" void kernel_launch(void* const* d_in, const int* in_sizes, int n_in,
                              void* d_out, int out_size, void* d_ws, size_t ws_size,
                              hipStream_t stream) {
  const float* x       = (const float*)d_in[0];
  const float* enc_W   = (const float*)d_in[1];
  const float* enc_U   = (const float*)d_in[2];
  const float* enc_b   = (const float*)d_in[3];
  const float* dec_W   = (const float*)d_in[4];
  const float* dec_U   = (const float*)d_in[5];
  const float* dec_b   = (const float*)d_in[6];
  const float* dense_W = (const float*)d_in[7];
  const float* dense_b = (const float*)d_in[8];
  float* out = (float*)d_out;
  float* hT  = (float*)d_ws;   // B*H f32 carry between kernels (128 KB)

  dim3 grid(B_SZ / 16), block(256);
  lstm_encoder_kernel<<<grid, block, 0, stream>>>(x, enc_W, enc_U, enc_b, hT);
  lstm_decoder_kernel<<<grid, block, 0, stream>>>(hT, dec_W, dec_U, dec_b, dense_W, dense_b, out);
}
